// NeuralPredicateLayer_76828374991788
// MI455X (gfx1250) — matrix operand order repaired and checked by reading, not compiled, not run on hardware
//
#include <hip/hip_runtime.h>
#include <math.h>

// ---------------------------------------------------------------------------
// NeuralPredicateLayer for MI455X (gfx1250): f16 WMMA (f32 accum) MLPs.
// Predicate index moved to blockIdx.y for occupancy (10k+ waves in flight).
// ---------------------------------------------------------------------------

typedef _Float16 h16;
typedef __attribute__((ext_vector_type(16))) _Float16 v16h;
typedef __attribute__((ext_vector_type(8)))  _Float16 v8h;
typedef __attribute__((ext_vector_type(8)))  float    v8f;

#define N_ENT   8192
#define M_PAIR  32768
#define F_DIM   256
#define H1      128
#define H2      64
#define PU_N    21
#define PB_N    6
#define BIN_K     544   // 516 padded up to 17 chunks of 32
#define BIN_K_RAW 516
#define HS_STRIDE (H1 + 8)   // 272 B rows -> 16B aligned, staggers LDS banks

// A-matrix fragment (16x32 f16, MxK): lane holds row m=lane&15.
// elems 0..7 -> K = kbase + (lane>>4)*8 + e ; elems 8..15 -> K += 16.
__device__ __forceinline__ v16h load_fragA(const h16* mat, int stride, int kbase, int lane) {
  const h16* rp = mat + (lane & 15) * stride + kbase + ((lane >> 4) << 3);
  v8h lo = *(const v8h*)rp;         // 16B load
  v8h hi = *(const v8h*)(rp + 16);  // 16B load
  v16h r;
#pragma unroll
  for (int i = 0; i < 8; ++i) { r[i] = lo[i]; r[i + 8] = hi[i]; }
  return r;
}

// B-matrix fragment (32x16 f16, KxN) with B[k][n] = W[n][k] (row-major W along K):
// lane holds col n=lane&15; elems 0..7 -> K = kbase + (lane>>4)*8 + e ; elems 8..15 -> K += 16.
__device__ __forceinline__ v16h load_fragB(const h16* mat, int stride, int kbase, int lane) {
  const h16* rp = mat + (lane & 15) * stride + kbase + ((lane >> 4) << 3);
  v8h lo = *(const v8h*)rp;
  v8h hi = *(const v8h*)(rp + 16);
  v16h r;
#pragma unroll
  for (int i = 0; i < 8; ++i) { r[i] = lo[i]; r[i + 8] = hi[i]; }
  return r;
}

__device__ __forceinline__ v8f splat8(float x) {
  v8f r;
#pragma unroll
  for (int i = 0; i < 8; ++i) r[i] = x;
  return r;
}

__device__ __forceinline__ float sigmoidf_(float x) { return 1.0f / (1.0f + __expf(-x)); }

// ---------------------------------------------------------------------------
// fp32 -> f16 conversion helpers (weights are tiny; features 8 MB)
// ---------------------------------------------------------------------------
__global__ __launch_bounds__(256) void cvt_f16_kernel(const float* __restrict__ src,
                                                      h16* __restrict__ dst, int n) {
  int i = blockIdx.x * blockDim.x + threadIdx.x;
  int stride = gridDim.x * blockDim.x;
  for (; i < n; i += stride) dst[i] = (h16)src[i];
}

// Wb1: (PB, H1, 516) -> (PB*H1, 544) zero-padded rows
__global__ __launch_bounds__(256) void cvt_pad_wb1_kernel(const float* __restrict__ src,
                                                          h16* __restrict__ dst) {
  int r = blockIdx.x;  // 0 .. PB*H1-1
  for (int k = threadIdx.x; k < BIN_K; k += blockDim.x)
    dst[r * BIN_K + k] = (k < BIN_K_RAW) ? (h16)src[r * BIN_K_RAW + k] : (h16)0.0f;
}

// ---------------------------------------------------------------------------
// Unary MLP: one wave per (16-row tile, predicate).
// ---------------------------------------------------------------------------
__global__ __launch_bounds__(32) void unary_mlp_kernel(
    const h16* __restrict__ ebf,
    const h16* __restrict__ W1, const float* __restrict__ b1,
    const h16* __restrict__ W2, const float* __restrict__ b2,
    const float* __restrict__ w3, const float* __restrict__ b3,
    float* __restrict__ out) {
  const int lane = threadIdx.x;
  const int row0 = blockIdx.x * 16;
  const int p = blockIdx.y;
  __shared__ h16 Hs[16][HS_STRIDE];

  // X tile fragments (16x256): 8 K-chunks, reused for all N-tiles
  v16h a1[F_DIM / 32];
#pragma unroll
  for (int c = 0; c < F_DIM / 32; ++c)
    a1[c] = load_fragA(ebf + row0 * F_DIM, F_DIM, c * 32, lane);

  const h16* W1p = W1 + p * H1 * F_DIM;
  const h16* W2p = W2 + p * H2 * H1;

  // ---- layer 1: (16x256)(256x128) ----
#pragma unroll
  for (int nt = 0; nt < H1 / 16; ++nt) {
    v8f acc = splat8(b1[p * H1 + nt * 16 + (lane & 15)]);
#pragma unroll
    for (int c = 0; c < F_DIM / 32; ++c) {
      v16h bfrag = load_fragB(W1p + nt * 16 * F_DIM, F_DIM, c * 32, lane);
      acc = __builtin_amdgcn_wmma_f32_16x16x32_f16(false, a1[c], false, bfrag,
                                                   (short)0, acc, false, false);
    }
#pragma unroll
    for (int r = 0; r < 8; ++r) {
      float v = acc[r] > 0.0f ? acc[r] : 0.0f;
      Hs[r + ((lane >> 4) << 3)][nt * 16 + (lane & 15)] = (h16)v;  // relu -> f16
    }
  }
  __syncthreads();

  // ---- layer 2: (16x128)(128x64), fold layer-3 dot in immediately ----
  v16h a2[H1 / 32];
#pragma unroll
  for (int c = 0; c < H1 / 32; ++c)
    a2[c] = load_fragA(&Hs[0][0], HS_STRIDE, c * 32, lane);

  float s[8];
#pragma unroll
  for (int r = 0; r < 8; ++r) s[r] = 0.0f;

#pragma unroll
  for (int nt = 0; nt < H2 / 16; ++nt) {
    v8f acc = splat8(b2[p * H2 + nt * 16 + (lane & 15)]);
#pragma unroll
    for (int c = 0; c < H1 / 32; ++c) {
      v16h bfrag = load_fragB(W2p + nt * 16 * H1, H1, c * 32, lane);
      acc = __builtin_amdgcn_wmma_f32_16x16x32_f16(false, a2[c], false, bfrag,
                                                   (short)0, acc, false, false);
    }
    float w = w3[p * H2 + nt * 16 + (lane & 15)];
#pragma unroll
    for (int r = 0; r < 8; ++r) {
      float v = acc[r] > 0.0f ? acc[r] : 0.0f;  // relu
      s[r] += v * w;                            // dot with w3
    }
  }
  // reduce over the 16-lane N group (lanes share the same 8 rows per half)
#pragma unroll
  for (int off = 1; off < 16; off <<= 1)
#pragma unroll
    for (int r = 0; r < 8; ++r) s[r] += __shfl_xor(s[r], off, 32);

  if ((lane & 15) == 0) {
    float bb = b3[p];
    int mb = (lane >> 4) << 3;
#pragma unroll
    for (int r = 0; r < 8; ++r)
      out[(row0 + mb + r) * PU_N + p] = sigmoidf_(s[r] + bb);
  }
}

// ---------------------------------------------------------------------------
// Binary MLP: one wave per (16-pair tile, predicate); combined features
// built in LDS (gather hits L2: entity table is only 4 MB in f16).
// ---------------------------------------------------------------------------
__global__ __launch_bounds__(32) void binary_mlp_kernel(
    const h16* __restrict__ ebf,
    const float* __restrict__ positions,
    const int* __restrict__ pair_idx,
    const h16* __restrict__ W1, const float* __restrict__ b1,
    const h16* __restrict__ W2, const float* __restrict__ b2,
    const float* __restrict__ w3, const float* __restrict__ b3,
    float* __restrict__ out) {
  const int lane = threadIdx.x;
  const int pr0 = blockIdx.x * 16;
  const int p = blockIdx.y;
  __shared__ h16 Cmb[16][BIN_K];        // 16 x 544 combined tile (17.4 KB)
  __shared__ h16 Hs[16][HS_STRIDE];
  __shared__ int sia[16], sib[16];

  if (lane < 16) {
    int ia = pair_idx[(pr0 + lane) * 2 + 0];
    int ib = pair_idx[(pr0 + lane) * 2 + 1];
    sia[lane] = ia; sib[lane] = ib;
    float dx = positions[ib * 3 + 0] - positions[ia * 3 + 0];
    float dy = positions[ib * 3 + 1] - positions[ia * 3 + 1];
    float dz = positions[ib * 3 + 2] - positions[ia * 3 + 2];
    float dist = sqrtf(dx * dx + dy * dy + dz * dz);
    Cmb[lane][512] = (h16)dx;
    Cmb[lane][513] = (h16)dy;
    Cmb[lane][514] = (h16)dz;
    Cmb[lane][515] = (h16)dist;
#pragma unroll
    for (int k = BIN_K_RAW; k < BIN_K; ++k) Cmb[lane][k] = (h16)0.0f;
  }
  __syncthreads();

  // gather fa|fb as 16-byte chunks: 16 rows x 64 chunks of 8 halves
  for (int c = lane; c < 16 * 64; c += 32) {
    int row = c >> 6;
    int cc = (c & 63) << 3;
    const h16* src = (cc < F_DIM) ? (ebf + sia[row] * F_DIM + cc)
                                  : (ebf + sib[row] * F_DIM + (cc - F_DIM));
    *(v8h*)&Cmb[row][cc] = *(const v8h*)src;
  }
  __syncthreads();

  // combined-tile A fragments (17 K-chunks) in registers
  v16h a1[BIN_K / 32];
#pragma unroll
  for (int c = 0; c < BIN_K / 32; ++c)
    a1[c] = load_fragA(&Cmb[0][0], BIN_K, c * 32, lane);

  const h16* W1p = W1 + p * H1 * BIN_K;
  const h16* W2p = W2 + p * H2 * H1;

  // ---- layer 1: (16x544)(544x128) ----
#pragma unroll
  for (int nt = 0; nt < H1 / 16; ++nt) {
    v8f acc = splat8(b1[p * H1 + nt * 16 + (lane & 15)]);
#pragma unroll
    for (int c = 0; c < BIN_K / 32; ++c) {
      v16h bfrag = load_fragB(W1p + nt * 16 * BIN_K, BIN_K, c * 32, lane);
      acc = __builtin_amdgcn_wmma_f32_16x16x32_f16(false, a1[c], false, bfrag,
                                                   (short)0, acc, false, false);
    }
#pragma unroll
    for (int r = 0; r < 8; ++r) {
      float v = acc[r] > 0.0f ? acc[r] : 0.0f;
      Hs[r + ((lane >> 4) << 3)][nt * 16 + (lane & 15)] = (h16)v;
    }
  }
  __syncthreads();

  // ---- layer 2 + fused layer-3 dot ----
  float s[8];
#pragma unroll
  for (int r = 0; r < 8; ++r) s[r] = 0.0f;

#pragma unroll
  for (int nt = 0; nt < H2 / 16; ++nt) {
    v8f acc = splat8(b2[p * H2 + nt * 16 + (lane & 15)]);
#pragma unroll
    for (int c = 0; c < H1 / 32; ++c) {
      v16h afrag = load_fragA(&Hs[0][0], HS_STRIDE, c * 32, lane);
      v16h bfrag = load_fragB(W2p + nt * 16 * H1, H1, c * 32, lane);
      acc = __builtin_amdgcn_wmma_f32_16x16x32_f16(false, afrag, false, bfrag,
                                                   (short)0, acc, false, false);
    }
    float w = w3[p * H2 + nt * 16 + (lane & 15)];
#pragma unroll
    for (int r = 0; r < 8; ++r) {
      float v = acc[r] > 0.0f ? acc[r] : 0.0f;
      s[r] += v * w;
    }
  }
#pragma unroll
  for (int off = 1; off < 16; off <<= 1)
#pragma unroll
    for (int r = 0; r < 8; ++r) s[r] += __shfl_xor(s[r], off, 32);

  if ((lane & 15) == 0) {
    float bb = b3[p];
    int mb = (lane >> 4) << 3;
#pragma unroll
    for (int r = 0; r < 8; ++r)
      out[(pr0 + mb + r) * PB_N + p] = sigmoidf_(s[r] + bb);
  }
}

// ---------------------------------------------------------------------------
// Launch
// ---------------------------------------------------------------------------
extern "C" void kernel_launch(void* const* d_in, const int* in_sizes, int n_in,
                              void* d_out, int out_size, void* d_ws, size_t ws_size,
                              hipStream_t stream) {
  const float* ent  = (const float*)d_in[0];   // (8192, 256)
  const float* pos  = (const float*)d_in[1];   // (8192, 3)
  const int*   pidx = (const int*)  d_in[2];   // (32768, 2)
  const float* Wu1  = (const float*)d_in[3];   // (21,128,256)
  const float* bu1  = (const float*)d_in[4];   // (21,128)
  const float* Wu2  = (const float*)d_in[5];   // (21,64,128)
  const float* bu2  = (const float*)d_in[6];   // (21,64)
  const float* wu3  = (const float*)d_in[7];   // (21,64)
  const float* bu3  = (const float*)d_in[8];   // (21,)
  const float* Wb1  = (const float*)d_in[9];   // (6,128,516)
  const float* bb1  = (const float*)d_in[10];  // (6,128)
  const float* Wb2  = (const float*)d_in[11];  // (6,64,128)
  const float* bb2  = (const float*)d_in[12];  // (6,64)
  const float* wb3  = (const float*)d_in[13];  // (6,64)
  const float* bb3  = (const float*)d_in[14];  // (6,)
  float* out = (float*)d_out;

  // workspace layout (f16), all 16B-aligned segments
  h16* ebf  = (h16*)d_ws;                       // 8192*256
  h16* wu1h = ebf  + N_ENT * F_DIM;             // 21*128*256
  h16* wu2h = wu1h + PU_N * H1 * F_DIM;         // 21*64*128
  h16* wb1h = wu2h + PU_N * H2 * H1;            // 6*128*544 (padded)
  h16* wb2h = wb1h + PB_N * H1 * BIN_K;         // 6*64*128

  cvt_f16_kernel<<<2048, 256, 0, stream>>>(ent, ebf, N_ENT * F_DIM);
  cvt_f16_kernel<<<1024, 256, 0, stream>>>(Wu1, wu1h, PU_N * H1 * F_DIM);
  cvt_f16_kernel<<<256,  256, 0, stream>>>(Wu2, wu2h, PU_N * H2 * H1);
  cvt_pad_wb1_kernel<<<PB_N * H1, 256, 0, stream>>>(Wb1, wb1h);
  cvt_f16_kernel<<<128,  256, 0, stream>>>(Wb2, wb2h, PB_N * H2 * H1);

  unary_mlp_kernel<<<dim3(N_ENT / 16, PU_N), 32, 0, stream>>>(
      ebf, wu1h, bu1, wu2h, bu2, wu3, bu3, out);
  binary_mlp_kernel<<<dim3(M_PAIR / 16, PB_N), 32, 0, stream>>>(
      ebf, pos, pidx, wb1h, bb1, wb2h, bb2, wb3, bb3, out + N_ENT * PU_N);
}
